// GraphAttentionLayer_85890755985565
// MI455X (gfx1250) — matrix operand order repaired and checked by reading, not compiled, not run on hardware
//
#include <hip/hip_runtime.h>
#include <hip/hip_bf16.h>
#include <stdint.h>

#define N_NODES 4096
#define E_EDGES 16384
#define IN_FEAT 512
#define OUT_FEAT 64
#define HF 128   // concatenated width: [high | low]

typedef __attribute__((ext_vector_type(16))) __bf16 v16bf;
typedef __attribute__((ext_vector_type(8)))  __bf16 v8bf;
typedef __attribute__((ext_vector_type(8)))  float  v8f;
typedef __attribute__((ext_vector_type(4)))  float  v4f;

// native RNE converts (lower to v_cvt_*bf16* on gfx1250)
__device__ __forceinline__ __bf16 f2bf(float f) { return (__bf16)f; }

__device__ __forceinline__ unsigned pkbf(float a, float b) {
  unsigned short x = __builtin_bit_cast(unsigned short, (__bf16)a);
  unsigned short y = __builtin_bit_cast(unsigned short, (__bf16)b);
  return (unsigned)x | ((unsigned)y << 16);
}

__device__ __forceinline__ float leakyf(float v, float a) { return v > 0.f ? v : a * v; }

__device__ __forceinline__ float gamfn(float v) {
  float t = v + 3.f;
  t = t < 0.f ? 0.f : (t > 6.f ? 6.f : t);
  return t * (1.f / 3.f) + 1e-6f;
}

// --- gfx1250 async copy: global -> LDS, 16B per lane, ASYNCcnt-tracked -------
__device__ __forceinline__ void async_b128(unsigned lds_off, const void* gaddr) {
  asm volatile("global_load_async_to_lds_b128 %0, %1, off"
               :: "v"(lds_off), "v"((unsigned long long)(uintptr_t)gaddr)
               : "memory");
}
// non-temporal variant for the once-streamed 256MB matrix (don't thrash L2)
__device__ __forceinline__ void async_b128_nt(unsigned lds_off, const void* gaddr) {
  asm volatile("global_load_async_to_lds_b128 %0, %1, off th:TH_LOAD_NT"
               :: "v"(lds_off), "v"((unsigned long long)(uintptr_t)gaddr)
               : "memory");
}
__device__ __forceinline__ void wait_async_le4() {
  asm volatile("s_wait_asynccnt 0x4" ::: "memory");
}
__device__ __forceinline__ void wait_async_le0() {
  asm volatile("s_wait_asynccnt 0x0" ::: "memory");
}

// ---------------------------------------------------------------------------
// K1: h[N][128] = leaky( x[N][512] @ [W_high | W_low][512][128], 0.01 )
// 128x128 tile/block, 8 waves x (16 rows x 128 cols). W tile stored [n][k].
// ---------------------------------------------------------------------------
__global__ __launch_bounds__(256) void k1_proj(const float* __restrict__ x,
                                               const float* __restrict__ Wh,
                                               const float* __restrict__ Wl,
                                               float* __restrict__ h) {
  constexpr int BM = 128, KC = 32, LDA = KC + 8, LDW = KC + 8;
  __shared__ __align__(16) __bf16 Xs[BM * LDA];       // [row][k]
  __shared__ __align__(16) __bf16 Wt[HF * LDW];       // [n][k]
  const int t = threadIdx.x;
  const int lane = t & 31;
  const int w = t >> 5;
  const int r = lane & 15, hi = lane >> 4;
  const int rowBase = blockIdx.x * BM;
  const int mr = w * 16;

  v8f acc[8] = {};

  for (int kb = 0; kb < IN_FEAT; kb += KC) {
    { // x tile 128x32 f32 -> bf16 (row-major, packed pair stores)
      const int row = t >> 1, cb = (t & 1) * 16;
      const float* src = x + (size_t)(rowBase + row) * IN_FEAT + kb + cb;
      #pragma unroll
      for (int i = 0; i < 16; i += 4) {
        float4 v = *(const float4*)(src + i);
        unsigned* dp = (unsigned*)&Xs[row * LDA + cb + i];
        dp[0] = pkbf(v.x, v.y);
        dp[1] = pkbf(v.z, v.w);
      }
    }
    { // W tile 32x128 -> transposed LDS [n][k]
      const int k = t >> 3, nb = (t & 7) * 16;
      const float* src = (nb < OUT_FEAT)
                             ? (Wh + (size_t)(kb + k) * OUT_FEAT + nb)
                             : (Wl + (size_t)(kb + k) * OUT_FEAT + (nb - OUT_FEAT));
      #pragma unroll
      for (int i = 0; i < 16; i += 4) {
        float4 v = *(const float4*)(src + i);
        Wt[(nb + i + 0) * LDW + k] = f2bf(v.x);
        Wt[(nb + i + 1) * LDW + k] = f2bf(v.y);
        Wt[(nb + i + 2) * LDW + k] = f2bf(v.z);
        Wt[(nb + i + 3) * LDW + k] = f2bf(v.w);
      }
    }
    __syncthreads();

    // A fragment: lane row mr+r; k = 16hi..16hi+15 contiguous (2x b128)
    const __bf16* pa = &Xs[(mr + r) * LDA];
    v8bf alo = *(const v8bf*)(pa + 16 * hi);
    v8bf ahi = *(const v8bf*)(pa + 16 * hi + 8);
    v16bf a = __builtin_shufflevector(alo, ahi, 0,1,2,3,4,5,6,7,8,9,10,11,12,13,14,15);

    #pragma unroll
    for (int nt = 0; nt < 8; ++nt) {
      // B fragment: column nt*16+r, k = 16hi..16hi+15 contiguous (2x b128)
      const __bf16* pb = &Wt[(nt * 16 + r) * LDW + 16 * hi];
      v8bf blo = *(const v8bf*)(pb);
      v8bf bhi = *(const v8bf*)(pb + 8);
      v16bf b = __builtin_shufflevector(blo, bhi, 0,1,2,3,4,5,6,7,8,9,10,11,12,13,14,15);
      acc[nt] = __builtin_amdgcn_wmma_f32_16x16x32_bf16(false, a, false, b,
                                                        (short)0, acc[nt], false, false);
    }
    __syncthreads();
  }

  #pragma unroll
  for (int nt = 0; nt < 8; ++nt) {
    #pragma unroll
    for (int v = 0; v < 8; ++v) {
      int mloc = v + 8 * hi;
      int col = nt * 16 + r;
      h[(size_t)(rowBase + mr + mloc) * HF + col] = leakyf(acc[nt][v], 0.01f);
    }
  }
}

// ---------------------------------------------------------------------------
// K2: per-edge (one wave32 per edge): in1|in2 -> inB (row-major bf16 [E][128]),
// attention scores, edge_e = exp(-leaky(s,0.2)), atomic row sums.
// ---------------------------------------------------------------------------
__global__ __launch_bounds__(256) void k2_edges(const float* __restrict__ h,
                                                const int* __restrict__ edge,
                                                const float* __restrict__ aH,
                                                const float* __restrict__ aL,
                                                const float* __restrict__ gH,
                                                const float* __restrict__ bHv,
                                                const float* __restrict__ gL,
                                                const float* __restrict__ bLv,
                                                __bf16* __restrict__ inB,
                                                float* __restrict__ ee,
                                                float* __restrict__ rs) {
  const int w = threadIdx.x >> 5, lane = threadIdx.x & 31;
  const int e = blockIdx.x * 8 + w;
  const int s = edge[e], d = edge[E_EDGES + e];

  const float gamma_h = gamfn(gH[0]), beta_h = gamfn(bHv[0]);
  const float gamma_l = gamfn(gL[0]), beta_l = gamfn(bLv[0]);

  float ph = 0.f, pl = 0.f, nh = 0.f, nl = 0.f;
  #pragma unroll
  for (int q = 0; q < 2; ++q) {
    const int f = lane + q * 32;
    float hhs = h[(size_t)s * HF + f];
    float hhd = h[(size_t)d * HF + f];
    float hls = h[(size_t)s * HF + 64 + f];
    float hld = h[(size_t)d * HF + 64 + f];
    float in1 = hhs - hhd;
    float in2 = hls + hld;
    inB[(size_t)e * HF + f]      = f2bf(in1);
    inB[(size_t)e * HF + 64 + f] = f2bf(in2);
    float a0 = aH[f], a1 = aH[64 + f], a2 = aH[128 + f], a3 = aH[192 + f];
    ph += beta_h * hhs * a0 + (2.f - beta_h) * hhd * a1
        + gamma_h * in1 * a2 + (2.f - gamma_h) * in2 * a3;
    nh += a0 * a0 + a1 * a1 + a2 * a2 + a3 * a3;
    float c0 = aL[f], c1 = aL[64 + f], c2 = aL[128 + f], c3 = aL[192 + f];
    pl += beta_l * hls * c0 + (2.f - beta_l) * hld * c1
        + gamma_l * in1 * c2 + (2.f - gamma_l) * in2 * c3;
    nl += c0 * c0 + c1 * c1 + c2 * c2 + c3 * c3;
  }
  #pragma unroll
  for (int off = 16; off > 0; off >>= 1) {
    ph += __shfl_xor(ph, off, 32);
    pl += __shfl_xor(pl, off, 32);
    nh += __shfl_xor(nh, off, 32);
    nl += __shfl_xor(nl, off, 32);
  }
  if (lane == 0) {
    float sH = ph * rsqrtf(nh);
    float sL = pl * rsqrtf(nl);
    float eH = __expf(-leakyf(sH, 0.2f));
    float eL = __expf(-leakyf(sL, 0.2f));
    ee[e * 2 + 0] = eH;
    ee[e * 2 + 1] = eL;
    atomicAdd(&rs[s * 2 + 0], eH);
    atomicAdd(&rs[s * 2 + 1], eL);
  }
}

// ---------------------------------------------------------------------------
// K2T: transpose inB [E][128] -> inBt [128][E] (bf16), 32x32 LDS tiles.
// ---------------------------------------------------------------------------
__global__ __launch_bounds__(256) void k2_transpose(const __bf16* __restrict__ in,
                                                    __bf16* __restrict__ outp) {
  __shared__ __bf16 tile[32][33];
  const int e0 = blockIdx.x * 32, f0 = blockIdx.y * 32;
  const int tx = threadIdx.x & 31, ty = threadIdx.x >> 5;   // 32 x 8
  #pragma unroll
  for (int i = ty; i < 32; i += 8)
    tile[i][tx] = in[(size_t)(e0 + i) * HF + f0 + tx];
  __syncthreads();
  #pragma unroll
  for (int i = ty; i < 32; i += 8)
    outp[(size_t)(f0 + i) * E_EDGES + e0 + tx] = tile[tx][i];
}

// ---------------------------------------------------------------------------
// K3: h2p[ks][N][128] partial of m[N][16384] @ inBt^T, K split 4 ways.
// 64x128 tile/block; async double-buffered global->LDS (b128, ASYNCcnt),
// m streamed with TH_LOAD_NT; f32->bf16 conversion at A-fragment build.
// ---------------------------------------------------------------------------
__global__ __launch_bounds__(256) void k3_gemm(const float* __restrict__ m,
                                               const __bf16* __restrict__ inBt,
                                               float* __restrict__ h2p) {
  constexpr int BM = 64, KC = 32;
  constexpr int LAF = KC + 4;      // f32 stride 36 -> 144B (16B multiple)
  constexpr int LBT = KC + 8;      // bf16 stride 40 -> 80B (16B multiple)
  __shared__ __align__(16) float  Af[2][BM * LAF];   // [row][k] f32
  __shared__ __align__(16) __bf16 Bt[2][HF * LBT];   // [n][k]  bf16

  const int t = threadIdx.x;
  const int lane = t & 31, w = t >> 5;
  const int r = lane & 15, hi = lane >> 4;
  const int wr = w >> 1, wc = w & 1;
  const int rowBase = blockIdx.x * BM;
  const int ks = blockIdx.y;                 // K split index
  const int kBase = ks * (E_EDGES / 4);
  constexpr int NCH = (E_EDGES / 4) / KC;    // 128 chunks
  const int mr = wr * 16;
  const int nc0 = wc * 64;

  const unsigned afOff0 = (unsigned)(uintptr_t)&Af[0][0];
  const unsigned afOff1 = (unsigned)(uintptr_t)&Af[1][0];
  const unsigned btOff0 = (unsigned)(uintptr_t)&Bt[0][0];
  const unsigned btOff1 = (unsigned)(uintptr_t)&Bt[1][0];

  // per-thread async-copy assignment (4 x b128 per chunk)
  const int arow = t >> 2, aq = t & 3;       // A: row 0..63, quarters {aq, aq+4}
  const int bcol = t >> 1, bh = t & 1;       // B: col 0..127, halves {bh, bh+2}

  v8f acc[4] = {};

  auto issue = [&](int kb, unsigned afOff, unsigned btOff) {
    const float* ga = m + (size_t)(rowBase + arow) * E_EDGES + kb;
    async_b128_nt(afOff + (unsigned)(arow * LAF + aq * 4) * 4u,       ga + aq * 4);
    async_b128_nt(afOff + (unsigned)(arow * LAF + (aq + 4) * 4) * 4u, ga + (aq + 4) * 4);
    const __bf16* gb = inBt + (size_t)bcol * E_EDGES + kb;
    async_b128(btOff + (unsigned)(bcol * LBT + bh * 8) * 2u,       gb + bh * 8);
    async_b128(btOff + (unsigned)(bcol * LBT + (bh + 2) * 8) * 2u, gb + (bh + 2) * 8);
  };

  auto compute = [&](const float* af, const __bf16* bt) {
    const float* pa = af + (mr + r) * LAF;
    v4f la0 = *(const v4f*)(pa + 16 * hi);
    v4f la1 = *(const v4f*)(pa + 16 * hi + 4);
    v4f la2 = *(const v4f*)(pa + 16 * hi + 8);
    v4f la3 = *(const v4f*)(pa + 16 * hi + 12);
    v16bf a{};
    #pragma unroll
    for (int j = 0; j < 4; ++j) {
      a[j]      = (__bf16)la0[j];
      a[4 + j]  = (__bf16)la1[j];
      a[8 + j]  = (__bf16)la2[j];
      a[12 + j] = (__bf16)la3[j];
    }
    #pragma unroll
    for (int nt = 0; nt < 4; ++nt) {
      const __bf16* pb = bt + (nc0 + nt * 16 + r) * LBT + 16 * hi;
      v8bf blo = *(const v8bf*)(pb);
      v8bf bhi8 = *(const v8bf*)(pb + 8);
      v16bf b = __builtin_shufflevector(blo, bhi8, 0,1,2,3,4,5,6,7,8,9,10,11,12,13,14,15);
      acc[nt] = __builtin_amdgcn_wmma_f32_16x16x32_bf16(false, a, false, b,
                                                        (short)0, acc[nt], false, false);
    }
  };

  issue(kBase, afOff0, btOff0);                       // chunk 0 -> buf0
  for (int c = 0; c < NCH; c += 2) {
    const int kb = kBase + c * KC;
    // L2 warm-up well ahead of the async engine
    if (c + 8 < NCH)
      __builtin_prefetch(m + (size_t)(rowBase + arow) * E_EDGES + kb + 8 * KC, 0, 1);

    if (c + 1 < NCH) { issue(kb + KC, afOff1, btOff1); wait_async_le4(); }
    else             { wait_async_le0(); }
    __syncthreads();
    compute(&Af[0][0], &Bt[0][0]);
    __syncthreads();

    if (c + 2 < NCH) { issue(kb + 2 * KC, afOff0, btOff0); wait_async_le4(); }
    else             { wait_async_le0(); }
    __syncthreads();
    compute(&Af[1][0], &Bt[1][0]);
    __syncthreads();
  }

  float* outp = h2p + (size_t)ks * N_NODES * HF;
  #pragma unroll
  for (int nt = 0; nt < 4; ++nt) {
    #pragma unroll
    for (int v = 0; v < 8; ++v) {
      outp[(size_t)(rowBase + mr + v + 8 * hi) * HF + nc0 + nt * 16 + r] = acc[nt][v];
    }
  }
}

// ---------------------------------------------------------------------------
// K3b: h2 = sum of 4 K-split partials (deterministic).
// ---------------------------------------------------------------------------
__global__ __launch_bounds__(256) void k3_reduce(const float* __restrict__ p,
                                                 float* __restrict__ h2) {
  const size_t S = (size_t)N_NODES * HF;
  size_t i = (size_t)blockIdx.x * 256 + threadIdx.x;
  h2[i] = p[i] + p[S + i] + p[2 * S + i] + p[3 * S + i];
}

// ---------------------------------------------------------------------------
// K4: hp[src] += edge_e * h2[dst]  (f32 global atomics, high|low halves)
// ---------------------------------------------------------------------------
__global__ __launch_bounds__(256) void k4_scatter(const int* __restrict__ edge,
                                                  const float* __restrict__ ee,
                                                  const float* __restrict__ h2,
                                                  float* __restrict__ hp) {
  int tg = blockIdx.x * 256 + threadIdx.x;
  int e = tg >> 7, f = tg & 127;
  int s = edge[e], d = edge[E_EDGES + e];
  float coef = ee[e * 2 + (f >> 6)];
  float val = coef * h2[(size_t)d * HF + f];
  atomicAdd(&hp[(size_t)s * HF + f], val);
}

// ---------------------------------------------------------------------------
// K5: out = leaky( hp_high/(rs_high+1e-16) + hp_low/(rs_low+1e-16), 0.01 )
// ---------------------------------------------------------------------------
__global__ __launch_bounds__(256) void k5_final(const float* __restrict__ hp,
                                                const float* __restrict__ rs,
                                                float* __restrict__ out) {
  int tg = blockIdx.x * 256 + threadIdx.x;
  int i = tg >> 6, f = tg & 63;
  float v = hp[(size_t)i * HF + f]      / (rs[i * 2 + 0] + 1e-16f)
          + hp[(size_t)i * HF + 64 + f] / (rs[i * 2 + 1] + 1e-16f);
  out[(size_t)i * OUT_FEAT + f] = leakyf(v, 0.01f);
}

// ---------------------------------------------------------------------------
extern "C" void kernel_launch(void* const* d_in, const int* in_sizes, int n_in,
                              void* d_out, int out_size, void* d_ws, size_t ws_size,
                              hipStream_t stream) {
  const float* x   = (const float*)d_in[0];
  const float* m   = (const float*)d_in[1];
  const float* Wh  = (const float*)d_in[2];
  const float* Wl  = (const float*)d_in[3];
  const float* aH  = (const float*)d_in[4];
  const float* aL  = (const float*)d_in[5];
  const float* gH  = (const float*)d_in[6];
  const float* bH  = (const float*)d_in[7];
  const float* gL  = (const float*)d_in[8];
  const float* bL  = (const float*)d_in[9];
  const int*   edg = (const int*)d_in[10];
  float* out = (float*)d_out;

  char* ws = (char*)d_ws;
  size_t off = 0;
  float*  h    = (float*)(ws + off);  off += (size_t)N_NODES * HF * sizeof(float);
  __bf16* inB  = (__bf16*)(ws + off); off += (size_t)E_EDGES * HF * sizeof(__bf16);
  __bf16* inBt = (__bf16*)(ws + off); off += (size_t)E_EDGES * HF * sizeof(__bf16);
  float*  ee   = (float*)(ws + off);  off += (size_t)E_EDGES * 2 * sizeof(float);
  float*  rs   = (float*)(ws + off);  off += (size_t)N_NODES * 2 * sizeof(float);
  float*  h2   = (float*)(ws + off);  off += (size_t)N_NODES * HF * sizeof(float);
  float*  hp   = (float*)(ws + off);  off += (size_t)N_NODES * HF * sizeof(float);
  float*  h2p  = (float*)(ws + off);  off += (size_t)4 * N_NODES * HF * sizeof(float);
  (void)ws_size; (void)in_sizes; (void)n_in; (void)out_size;

  hipMemsetAsync(rs, 0, (size_t)N_NODES * 2 * sizeof(float), stream);
  hipMemsetAsync(hp, 0, (size_t)N_NODES * HF * sizeof(float), stream);

  k1_proj<<<N_NODES / 128, 256, 0, stream>>>(x, Wh, Wl, h);
  k2_edges<<<E_EDGES / 8, 256, 0, stream>>>(h, edg, aH, aL, gH, bH, gL, bL, inB, ee, rs);
  k2_transpose<<<dim3(E_EDGES / 32, HF / 32), 256, 0, stream>>>(inB, inBt);
  k3_gemm<<<dim3(N_NODES / 64, 4), 256, 0, stream>>>(m, inBt, h2p);
  k3_reduce<<<(N_NODES * HF) / 256, 256, 0, stream>>>(h2p, h2);
  k4_scatter<<<(E_EDGES * HF) / 256, 256, 0, stream>>>(edg, ee, h2, hp);
  k5_final<<<(N_NODES * OUT_FEAT) / 256, 256, 0, stream>>>(hp, rs, out);
}
